// FinalBlock_60790967108049
// MI455X (gfx1250) — matrix operand order repaired and checked by reading, not compiled, not run on hardware
//
#include <hip/hip_runtime.h>
#include <hip/hip_bf16.h>

#define L     1536
#define EPAIR 32
#define TOPK  64
#define KMIN_ 9

typedef __attribute__((ext_vector_type(16))) __bf16 v16bf;
typedef __attribute__((ext_vector_type(8)))  float  v8f;

// --------------------------------------------------------------------------
// CDNA5 async global->LDS path (ASYNCcnt), guarded so either toolchain builds.
// Builtin signature (from hipcc diagnostic): pointers to 16-byte int vectors,
// AS1 (global) source first, AS3 (LDS) dest second.
// --------------------------------------------------------------------------
#if __has_builtin(__builtin_amdgcn_global_load_async_to_lds_b128)
#define HAS_ASYNC_LDS 1
typedef int v4i_ __attribute__((vector_size(16)));
#define AS1(p) ((__attribute__((address_space(1))) v4i_*)(unsigned long long)(const void*)(p))
#define AS3(p) ((__attribute__((address_space(3))) v4i_*)(unsigned int)(unsigned long long)(void*)(p))
#if __has_builtin(__builtin_amdgcn_s_wait_asynccnt)
#define WAIT_ASYNC(n) __builtin_amdgcn_s_wait_asynccnt(n)
#else
#define WAIT_ASYNC(n) asm volatile("s_wait_asynccnt " #n ::: "memory")
#endif
#endif

// 16-lane (row) sum via DPP: xor1, xor2 (quad_perm), xor4 (row_half_mirror),
// xor8 (row_mirror).  Pure VALU; no LDS round-trips.
__device__ __forceinline__ float sum_half16(float x) {
  x += __int_as_float(__builtin_amdgcn_update_dpp(0, __float_as_int(x), 0xB1,  0xF, 0xF, true));
  x += __int_as_float(__builtin_amdgcn_update_dpp(0, __float_as_int(x), 0x4E,  0xF, 0xF, true));
  x += __int_as_float(__builtin_amdgcn_update_dpp(0, __float_as_int(x), 0x141, 0xF, 0xF, true));
  x += __int_as_float(__builtin_amdgcn_update_dpp(0, __float_as_int(x), 0x140, 0xF, 0xF, true));
  return x;
}

// ---------------------------------------------------------------------------
// Kernel A: nW[j,o] = node[j,:] @ W_n[:,o]   (tiny: 1536x32x32)
// ---------------------------------------------------------------------------
__global__ void node_proj_kernel(const float* __restrict__ node,
                                 const float* __restrict__ W_n,
                                 float* __restrict__ nW) {
  int j = blockIdx.x;
  int o = threadIdx.x;
  float acc = 0.f;
#pragma unroll
  for (int d = 0; d < 32; ++d) acc += node[j * 32 + d] * W_n[d * 32 + o];
  nW[j * 32 + o] = acc;
}

// ---------------------------------------------------------------------------
// Kernel B: per-row 64th-smallest distance (radix select on float bits in LDS)
//           + degree of (top64 OR seq-sep<9) mask
// ---------------------------------------------------------------------------
__global__ void thresh_kernel(const float* __restrict__ xyz,
                              const int* __restrict__ idx,
                              float* __restrict__ thr,
                              float* __restrict__ deg) {
  __shared__ unsigned int keys[L];
  __shared__ int cnt;
  int i = blockIdx.x;
  int tid = threadIdx.x;
  float cx = xyz[i * 9 + 3], cy = xyz[i * 9 + 4], cz = xyz[i * 9 + 5];
  for (int j = tid; j < L; j += blockDim.x) {
    float dx = xyz[j * 9 + 3] - cx;
    float dy = xyz[j * 9 + 4] - cy;
    float dz = xyz[j * 9 + 5] - cz;
    float D = sqrtf(dx * dx + dy * dy + dz * dz + 1e-12f);
    if (j == i) D += 999.9f;              // D_big diagonal
    keys[j] = __float_as_uint(D);         // positive floats: bit order == value order
  }
  __syncthreads();

  unsigned int prefix = 0;
  int k = TOPK;                            // rank (1-based) of element we seek
  for (int b = 31; b >= 0; --b) {
    if (tid == 0) cnt = 0;
    __syncthreads();
    unsigned int himask = (b == 31) ? 0u : (0xFFFFFFFFu << (b + 1));
    unsigned int bit = 1u << b;
    int local = 0;
    for (int j = tid; j < L; j += blockDim.x) {
      unsigned int u = keys[j];
      if ((u & himask) == prefix && (u & bit) == 0u) local++;
    }
    atomicAdd(&cnt, local);
    __syncthreads();
    int tot = cnt;
    __syncthreads();
    if (tot < k) { k -= tot; prefix |= bit; }   // uniform decision across block
  }
  if (tid == 0) thr[i] = __uint_as_float(prefix);

  if (tid == 0) cnt = 0;
  __syncthreads();
  int ii = idx[i];
  int local = 0;
  for (int j = tid; j < L; j += blockDim.x) {
    if (j == i) continue;
    int s = idx[j] - ii; if (s < 0) s = -s;
    if (keys[j] <= prefix || s < KMIN_) local++;
  }
  atomicAdd(&cnt, local);
  __syncthreads();
  if (tid == 0) deg[i] = (float)cnt;
}

// ---------------------------------------------------------------------------
// Kernel C: fused  relu(edge_feat@W_e + nW[j]) -> masked mean + gated vec sum
// One block per row i; 4 waves; j tiled by 16 (WMMA M); 32 outputs = 2 N-halves.
// K chunks: pair[0:32] | rbf[0:32] | rbf[32:36] (zero padded) -> 6 WMMAs/tile.
// pair stream is double-buffered in LDS via async global->LDS loads (ASYNCcnt).
// ---------------------------------------------------------------------------
__global__ __launch_bounds__(128, 1)
void fused_kernel(const float* __restrict__ xyz,
                  const float* __restrict__ pair,
                  const int* __restrict__ idx,
                  const float* __restrict__ W_e,
                  const float* __restrict__ w_gate,
                  const float* __restrict__ nW,
                  const float* __restrict__ thr,
                  const float* __restrict__ deg,
                  float* __restrict__ out) {
  __shared__ float sh[35];
#ifdef HAS_ASYNC_LDS
  __shared__ float pbuf[4][2][16 * EPAIR];   // per-wave double buffer, 16KB
#endif
  int i = blockIdx.x;
  int tid = threadIdx.x;
  if (tid < 35) sh[tid] = 0.f;
  __syncthreads();

  int lane = tid & 31;
  int wid  = tid >> 5;
  int n16  = lane & 15;
  int hi   = (lane >= 16) ? 1 : 0;
  int kb   = hi ? 8 : 0;                       // A-operand K base per half-wave

  float cx = xyz[i * 9 + 3], cy = xyz[i * 9 + 4], cz = xyz[i * 9 + 5];
  float thrF = thr[i];
  int   ii   = idx[i];
  float wg0 = w_gate[n16], wg1 = w_gate[16 + n16];

  // B operands in bf16 (B layout: e<8 -> K = 8*hi + e ; e>=8 -> K = 16 + 8*hi + (e-8), col = lane&15)
  v16bf B1[2], B2[2], B3[2];
#pragma unroll
  for (int h = 0; h < 2; ++h) {
#pragma unroll
    for (int e = 0; e < 16; ++e) {
      int kk = (e < 8) ? (8 * hi + e) : (16 + 8 * hi + (e - 8));
      B1[h][e] = (__bf16)W_e[kk * 32 + 16 * h + n16];
      B2[h][e] = (__bf16)W_e[(32 + kk) * 32 + 16 * h + n16];
      B3[h][e] = (kk < 4) ? (__bf16)W_e[(64 + kk) * 32 + 16 * h + n16]
                          : (__bf16)0.0f;
    }
  }

  float nodeacc0 = 0.f, nodeacc1 = 0.f;
  float vx = 0.f, vy = 0.f, vz = 0.f;

#ifdef HAS_ASYNC_LDS
  // async copy of this wave's first pair tile (16 rows x 32 f32 = 2KB):
  // 4 fully-coalesced 512B issues (one b128 per lane).
  {
    const float* gsrc = pair + ((size_t)i * L + (size_t)16 * wid) * EPAIR;
    float* ldst = &pbuf[wid][0][0];
#pragma unroll
    for (int c = 0; c < 4; ++c)
      __builtin_amdgcn_global_load_async_to_lds_b128(
          AS1(gsrc + lane * 4 + c * 128), AS3(ldst + lane * 4 + c * 128), 0, 0);
  }
#endif

  for (int it = 0; it < (L / 16) / 4; ++it) {  // 24 tiles per wave, no divergence
    int t  = wid + it * 4;
    int j0 = t * 16;
    int j  = j0 + n16;                         // this lane's A row (m = lane&15)

    float dx = xyz[j * 9 + 3] - cx;
    float dy = xyz[j * 9 + 4] - cy;
    float dz = xyz[j * 9 + 5] - cz;
    float Dv = sqrtf(dx * dx + dy * dy + dz * dz + 1e-12f);

#ifdef HAS_ASYNC_LDS
    // prefetch next tile into the other buffer, then wait for current buffer
    if (it + 1 < 24) {
      const float* gsrc = pair + ((size_t)i * L + (size_t)16 * (t + 4)) * EPAIR;
      float* ldst = &pbuf[wid][(it + 1) & 1][0];
#pragma unroll
      for (int c = 0; c < 4; ++c)
        __builtin_amdgcn_global_load_async_to_lds_b128(
            AS1(gsrc + lane * 4 + c * 128), AS3(ldst + lane * 4 + c * 128), 0, 0);
      WAIT_ASYNC(4);
    } else {
      WAIT_ASYNC(0);
    }
    const float* prow = &pbuf[wid][it & 1][n16 * EPAIR];
#else
    if (it + 1 < 24)
      __builtin_prefetch(pair + ((size_t)i * L + j + 64) * EPAIR, 0, 1);
    const float* prow = pair + ((size_t)i * L + j) * EPAIR;
#endif

    // A1: pair features.  A layout: e<8 -> K=kb+e ; e>=8 -> K=16+kb+(e-8)
    float4 p0 = *(const float4*)(prow + kb);
    float4 p1 = *(const float4*)(prow + kb + 4);
    float4 p2 = *(const float4*)(prow + 16 + kb);
    float4 p3 = *(const float4*)(prow + 16 + kb + 4);
    v16bf a1;
    a1[0]  = (__bf16)p0.x; a1[1]  = (__bf16)p0.y; a1[2]  = (__bf16)p0.z; a1[3]  = (__bf16)p0.w;
    a1[4]  = (__bf16)p1.x; a1[5]  = (__bf16)p1.y; a1[6]  = (__bf16)p1.z; a1[7]  = (__bf16)p1.w;
    a1[8]  = (__bf16)p2.x; a1[9]  = (__bf16)p2.y; a1[10] = (__bf16)p2.z; a1[11] = (__bf16)p2.w;
    a1[12] = (__bf16)p3.x; a1[13] = (__bf16)p3.y; a1[14] = (__bf16)p3.z; a1[15] = (__bf16)p3.w;

    // A2/A3: RBF features, branchless: rbf_k(D) = exp(-(D*1.8 - k*36/35)^2),
    // fold the per-half K base into Ds2 once; element offsets become literals.
    float Ds2 = Dv * 1.8f - (float)kb * (36.0f / 35.0f);
    v16bf a2, a3;
#pragma unroll
    for (int e = 0; e < 16; ++e) {
      const int eo = (e < 8) ? e : (16 + (e - 8));
      float t2 = Ds2 - (float)eo * (36.0f / 35.0f);
      a2[e] = (__bf16)__expf(-t2 * t2);
      if (e < 4) {                    // rbf feature 32+kk valid only when kb==0
        float t3 = Ds2 - (float)(32 + e) * (36.0f / 35.0f);
        float v3 = hi ? 0.f : __expf(-t3 * t3);
        a3[e] = (__bf16)v3;
      } else {
        a3[e] = (__bf16)0.0f;
      }
    }

    // C init = node bias nW[j, o]   (C layout: VGPR r -> M = r + 8*hi, N = lane&15)
    v8f c0, c1;
#pragma unroll
    for (int r = 0; r < 8; ++r) {
      int row = j0 + r + 8 * hi;
      c0[r] = nW[row * 32 + n16];
      c1[r] = nW[row * 32 + 16 + n16];
    }

    c0 = __builtin_amdgcn_wmma_f32_16x16x32_bf16(false, a1, false, B1[0], (short)0, c0, false, false);
    c0 = __builtin_amdgcn_wmma_f32_16x16x32_bf16(false, a2, false, B2[0], (short)0, c0, false, false);
    c0 = __builtin_amdgcn_wmma_f32_16x16x32_bf16(false, a3, false, B3[0], (short)0, c0, false, false);
    c1 = __builtin_amdgcn_wmma_f32_16x16x32_bf16(false, a1, false, B1[1], (short)0, c1, false, false);
    c1 = __builtin_amdgcn_wmma_f32_16x16x32_bf16(false, a2, false, B2[1], (short)0, c1, false, false);
    c1 = __builtin_amdgcn_wmma_f32_16x16x32_bf16(false, a3, false, B3[1], (short)0, c1, false, false);

    // neighbor mask for this lane's row; share via ballot (bit l = mask of m=l&15)
    int s = idx[j] - ii; if (s < 0) s = -s;
    bool mk = (j != i) && (Dv <= thrF || s < KMIN_);
    unsigned int mbits = __builtin_amdgcn_ballot_w32(mk);

#pragma unroll
    for (int r = 0; r < 8; ++r) {
      int m = r + 8 * hi;
      float fm = ((mbits >> m) & 1u) ? 1.0f : 0.0f;
      float r0 = (c0[r] > 0.f) ? c0[r] : 0.f;
      float r1 = (c1[r] > 0.f) ? c1[r] : 0.f;
      nodeacc0 += fm * r0;                      // channel n16
      nodeacc1 += fm * r1;                      // channel 16+n16
      // gate[m] = sum_o msg[m,o]*w_gate[o]; DPP 16-lane reduce (no LDS)
      float g = sum_half16(fm * (r0 * wg0 + r1 * wg1));
      // exactly one lane per m has n16==m inside the computing half
      // (lanes 0-7 for m=0-7, lanes 24-31 for m=8-15): it owns dx/dy/dz of row m
      float pick = (n16 == m) ? g : 0.0f;
      vx = fmaf(pick, dx, vx);
      vy = fmaf(pick, dy, vy);
      vz = fmaf(pick, dz, vz);
    }
  }

  atomicAdd(&sh[n16], nodeacc0);
  atomicAdd(&sh[16 + n16], nodeacc1);
  if (lane < 8 || lane >= 24) {               // only owner lanes hold nonzero v*
    atomicAdd(&sh[32], vx);
    atomicAdd(&sh[33], vy);
    atomicAdd(&sh[34], vz);
  }
  __syncthreads();

  if (tid < 35) {
    float inv = 1.0f / (deg[i] + 1e-6f);
    out[i * 35 + tid] = sh[tid] * inv;
  }
}

// ---------------------------------------------------------------------------
extern "C" void kernel_launch(void* const* d_in, const int* in_sizes, int n_in,
                              void* d_out, int out_size, void* d_ws, size_t ws_size,
                              hipStream_t stream) {
  const float* xyz   = (const float*)d_in[0];
  const float* pair  = (const float*)d_in[1];
  const float* node  = (const float*)d_in[2];
  const int*   idx   = (const int*)d_in[3];
  const float* W_e   = (const float*)d_in[4];
  const float* W_n   = (const float*)d_in[5];
  const float* wgate = (const float*)d_in[6];
  float* out = (float*)d_out;

  float* nW  = (float*)d_ws;                 // L*32 floats
  float* thr = nW + (size_t)L * 32;          // L floats
  float* deg = thr + L;                      // L floats

  node_proj_kernel<<<L, 32, 0, stream>>>(node, W_n, nW);
  thresh_kernel<<<L, 256, 0, stream>>>(xyz, idx, thr, deg);
  fused_kernel<<<L, 128, 0, stream>>>(xyz, pair, idx, W_e, wgate, nW, thr, deg, out);
}
